// BaseConchRd_16406775071375
// MI455X (gfx1250) — compile-verified, matrix operand order checked by reading, not run on hardware
//
#include <hip/hip_runtime.h>
#include <cstdint>
#include <math.h>

// ---------------- problem constants (match reference) ----------------
#define N_  65536
#define E_  262144
#define K_  8
#define NMP_ 2
#define DEPTH_ 2
#define H_  4
#define D_  256
#define PH_ 64
#define ED_ 64

typedef __attribute__((ext_vector_type(16))) _Float16 v16h;
typedef __attribute__((ext_vector_type(8)))  float    v8f;

// ---------------- CDNA5 async global->LDS (ASYNCcnt-tracked) ----------------
// Moves 16 bytes per lane from global memory straight into LDS without a VGPR
// round-trip. VDST VGPR holds the LDS byte address (low 32 bits of the flat
// LDS address per ISA aperture rule), VADDR holds the 64-bit global address.
static __device__ __forceinline__ void async_b128(void* lds_dst, const void* gsrc) {
  uint32_t ldsoff = (uint32_t)(uintptr_t)lds_dst;
  asm volatile("global_load_async_to_lds_b128 %0, %1, off"
               :: "v"(ldsoff), "v"(gsrc)
               : "memory");
}
static __device__ __forceinline__ void wait_async0() {
  asm volatile("s_wait_asynccnt 0x0" ::: "memory");
}

// ---------------- WMMA helpers (gfx1250, wave32) ----------------
static __device__ __forceinline__ v8f wmma16(v16h a, v16h b, v8f c) {
  // D = A(16x32 f16) * B(32x16 f16) + C(16x16 f32)
  return __builtin_amdgcn_wmma_f32_16x16x32_f16(
      /*neg_a=*/false, a, /*neg_b=*/false, b,
      /*c_mod=*/(short)0, c, /*reuse_a=*/false, /*reuse_b=*/false);
}

// A fragment from an LDS f16 tile (row-major, leading dim ld).
// ISA 7.12.2: lanes 0-15 hold M=lane, K = k0+{0..7,16..23}; lanes 16-31 K+8.
static __device__ __forceinline__ v16h load_a_lds_f16(const _Float16* tile, int ld,
                                                      int k0, int lane) {
  int m = lane & 15;
  int half = lane >> 4;
  const _Float16* p = tile + m * ld + k0 + half * 8;
  v16h a;
#pragma unroll
  for (int j = 0; j < 8; ++j) { a[j] = p[j]; a[8 + j] = p[16 + j]; }
  return a;
}

// A fragment from an LDS f32 tile, converting f32->f16 during the load.
static __device__ __forceinline__ v16h load_a_lds_f32(const float* tile, int ld,
                                                      int k0, int lane) {
  int m = lane & 15;
  int half = lane >> 4;
  const float* p = tile + m * ld + k0 + half * 8;
  v16h a;
#pragma unroll
  for (int j = 0; j < 8; ++j) { a[j] = (_Float16)p[j]; a[8 + j] = (_Float16)p[16 + j]; }
  return a;
}

// B fragment ([K x N] row-major) from an LDS f32 tile, f32->f16 on load.
static __device__ __forceinline__ v16h load_b_lds_f32(const float* B, int ldb,
                                                      int k0, int n0, int lane) {
  int n = lane & 15;
  int half = lane >> 4;
  const float* p = B + (size_t)(k0 + half * 8) * ldb + n0 + n;
  v16h b;
#pragma unroll
  for (int j = 0; j < 8; ++j) {
    b[j]     = (_Float16)p[(size_t)j * ldb];
    b[8 + j] = (_Float16)p[(size_t)(16 + j) * ldb];
  }
  return b;
}

// B fragment from a global f32 matrix (weights are tiny and L2-resident).
static __device__ __forceinline__ v16h load_b_glb(const float* B, int ldb,
                                                  int k0, int n0, int lane) {
  int n = lane & 15;
  int half = lane >> 4;
  const float* p = B + (size_t)(k0 + half * 8) * ldb + n0 + n;
  v16h b;
#pragma unroll
  for (int j = 0; j < 8; ++j) {
    b[j]     = (_Float16)p[(size_t)j * ldb];
    b[8 + j] = (_Float16)p[(size_t)(16 + j) * ldb];
  }
  return b;
}

static __device__ __forceinline__ v8f zero8() {
  v8f z = {0.f, 0.f, 0.f, 0.f, 0.f, 0.f, 0.f, 0.f};
  return z;
}

static __device__ __forceinline__ float lrelu(float v) { return v > 0.f ? v : 0.2f * v; }
static __device__ __forceinline__ float eluf(float v)  { return v > 0.f ? v : __expf(v) - 1.f; }

// ====================================================================
// Generic tiled GEMM:  C[M,Nd] = A[M,Kd] @ B[Kd,Nd]  (f32 memory, f16 WMMA
// math, f32 accumulate). Block = 128 thr (4 waves); tile 16 rows x 64 cols,
// one 16x16 tile per wave. grid = (Nd/64, M/16). A and B tiles are staged
// into LDS with async global->LDS b128 transfers.
// ====================================================================
__global__ void __launch_bounds__(128)
gemm_f16_wmma(const float* __restrict__ A, int lda,
              const float* __restrict__ B, int ldb,
              float* __restrict__ C, int ldc, int Kd) {
  __shared__ float At[16 * 32];   // 2 KB
  __shared__ float Bt[32 * 64];   // 8 KB
  const int tid = threadIdx.x, wave = tid >> 5, lane = tid & 31;
  const int row0 = blockIdx.y * 16;
  const int colB = blockIdx.x * 64;

  v8f acc = zero8();
  for (int k0 = 0; k0 < Kd; k0 += 32) {
    {  // A tile: 16x32 f32, one b128 per thread
      int r = tid >> 3, c = (tid & 7) * 4;
      async_b128(&At[r * 32 + c], &A[(size_t)(row0 + r) * lda + k0 + c]);
    }
    {  // B tile: 32x64 f32, four b128 per thread
#pragma unroll
      for (int u = 0; u < 4; ++u) {
        int t = tid + u * 128;
        int r = t >> 4, c = (t & 15) * 4;
        async_b128(&Bt[r * 64 + c], &B[(size_t)(k0 + r) * ldb + colB + c]);
      }
    }
    if (k0 + 32 < Kd) {  // speculative prefetch of next A tile
      int r = tid >> 3;
      __builtin_prefetch(&A[(size_t)(row0 + r) * lda + k0 + 32], 0, 1);
    }
    wait_async0();
    __syncthreads();
    v16h a = load_a_lds_f32(At, 32, 0, lane);
    v16h b = load_b_lds_f32(Bt, 64, 0, wave * 16, lane);
    acc = wmma16(a, b, acc);
    __syncthreads();
  }
  // C/D layout: vgpr r -> (M=r, N=lane) lanes 0-15 ; (M=r+8, N=lane-16) 16-31
  const int n = lane & 15, mb = (lane >> 4) * 8;
#pragma unroll
  for (int r = 0; r < 8; ++r)
    C[(size_t)(row0 + mb + r) * ldc + blockIdx.x * 64 + wave * 16 + n] = acc[r];
}

// ====================================================================
// Fused edge aggregator: per 16-edge tile, async-gather 2 endpoint rows each,
// per-head attention over the 2 endpoints, z = edges + agg, then
// e_out[:, h*64:(h+1)*64] = elu(z @ W_e[h]).  128 thr = 4 waves (1/head).
// ====================================================================
__global__ void __launch_bounds__(128)
edge_attn_proj(const float* __restrict__ x,      // [N, 256] node feats
               const float* __restrict__ edges,  // [E, 256]
               const int*   __restrict__ ena,    // [E, 2]
               const float* __restrict__ a_n,    // [H, 256]
               const float* __restrict__ a_e,    // [H, 256]
               const float* __restrict__ W,      // [H, 256, 64]
               float* __restrict__ eout) {       // [E, 256]
  extern __shared__ char smem[];
  float*    ep  = (float*)smem;                 // [32][256] endpoint rows (e*2+q)
  float*    edt = ep + 32 * 256;                // [16][256] edge rows
  _Float16* zt  = (_Float16*)(edt + 16 * 256);  // [4][16][256] per-head z (f16)
  float*    sc  = (float*)(zt + 4 * 16 * 256);  // [16][4][3] scores

  const int tid = threadIdx.x, wave = tid >> 5, lane = tid & 31;
  const int e0 = blockIdx.x * 16;
  const int h = wave;

  {  // async-gather 32 endpoint rows: 4 threads/row, 64 f32 (16 b128) each
    int r = tid >> 2, part = tid & 3;
    int e = r >> 1, q = r & 1;
    int node = ena[(size_t)(e0 + e) * 2 + q];
    const float* src = x + (size_t)node * 256 + part * 64;
    float* dst = ep + r * 256 + part * 64;
#pragma unroll
    for (int i = 0; i < 16; ++i) async_b128(dst + i * 4, src + i * 4);
  }
  {  // async-load 16 edge rows: 8 threads/row, 32 f32 (8 b128) each
    int r = tid >> 3, part = tid & 7;
    const float* src = edges + (size_t)(e0 + r) * 256 + part * 32;
    float* dst = edt + r * 256 + part * 32;
#pragma unroll
    for (int i = 0; i < 8; ++i) async_b128(dst + i * 4, src + i * 4);
  }
  wait_async0();
  __syncthreads();

  {  // endpoint attention dots: 32 tasks / wave
    const float* an = a_n + h * 256;
    int e = lane >> 1, q = lane & 1;
    const float* row = ep + (e * 2 + q) * 256;
    float s = 0.f;
    for (int d = 0; d < 256; ++d) s += row[d] * an[d];
    sc[(e * 4 + h) * 3 + q] = s;
  }
  if (lane < 16) {  // edge dots
    const float* ae = a_e + h * 256;
    const float* row = edt + lane * 256;
    float s = 0.f;
    for (int d = 0; d < 256; ++d) s += row[d] * ae[d];
    sc[(lane * 4 + h) * 3 + 2] = s;
  }
  __syncthreads();

  // softmax over the 2 endpoints, build z = edges + a0*ep0 + a1*ep1 (f16)
  for (int e = 0; e < 16; ++e) {
    float se = sc[(e * 4 + h) * 3 + 2];
    float l0 = lrelu(sc[(e * 4 + h) * 3 + 0] + se);
    float l1 = lrelu(sc[(e * 4 + h) * 3 + 1] + se);
    float m = fmaxf(l0, l1);
    float w0 = __expf(l0 - m), w1 = __expf(l1 - m);
    float inv = 1.f / (w0 + w1);
    w0 *= inv; w1 *= inv;
    const float* p0 = ep + (e * 2 + 0) * 256;
    const float* p1 = ep + (e * 2 + 1) * 256;
    const float* pe = edt + e * 256;
    _Float16* zd = zt + (h * 16 + e) * 256;
#pragma unroll
    for (int j = 0; j < 8; ++j) {
      int d = lane + 32 * j;
      zd[d] = (_Float16)(pe[d] + w0 * p0[d] + w1 * p1[d]);
    }
  }
  __syncthreads();

  // per-head projection: [16,256] x [256,64] via 4 N-tiles x 8 K-steps
  const float* Wh = W + (size_t)h * 256 * 64;
  const _Float16* zh = zt + h * 16 * 256;
  v8f a0 = zero8(), a1 = zero8(), a2 = zero8(), a3 = zero8();
  for (int k0 = 0; k0 < 256; k0 += 32) {
    v16h a = load_a_lds_f16(zh, 256, k0, lane);
    a0 = wmma16(a, load_b_glb(Wh, 64, k0,  0, lane), a0);
    a1 = wmma16(a, load_b_glb(Wh, 64, k0, 16, lane), a1);
    a2 = wmma16(a, load_b_glb(Wh, 64, k0, 32, lane), a2);
    a3 = wmma16(a, load_b_glb(Wh, 64, k0, 48, lane), a3);
  }
  const int n = lane & 15, mb = (lane >> 4) * 8;
#pragma unroll
  for (int r = 0; r < 8; ++r) {
    float* orow = eout + (size_t)(e0 + mb + r) * 256 + h * 64 + n;
    orow[0]  = eluf(a0[r]);
    orow[16] = eluf(a1[r]);
    orow[32] = eluf(a2[r]);
    orow[48] = eluf(a3[r]);
  }
}

// ====================================================================
// Fused node aggregator: per 16-node tile, async-gather K=8 edge rows each,
// per-head softmax over K, nag = sum_k alpha_k * eg_k, then
// n_out[:, h*64:(h+1)*64] = elu(xin @ W_self[h] + nag @ W_nb[h]).
// 128 thr = 4 waves (1/head). Output row stride ldo (skip-concat layout).
// ====================================================================
__global__ void __launch_bounds__(128)
node_attn_proj(const float* __restrict__ xin, int ldx,   // [N, ldx]
               const float* __restrict__ edges,          // [E, 256]
               const int*   __restrict__ n2e,            // [N, 8]
               const float* __restrict__ a_s,            // [H, 256]
               const float* __restrict__ a_ed,           // [H, 256]
               const float* __restrict__ Wself,          // [H, 256, 64]
               const float* __restrict__ Wnb,            // [H, 256, 64]
               float* __restrict__ out, int ldo) {
  extern __shared__ char smem[];
  float*    eg = (float*)smem;                  // [16*8][256] gathered edge rows
  float*    xf = eg + 16 * 8 * 256;             // [16][256] xin (f32)
  _Float16* xa = (_Float16*)(xf + 16 * 256);    // [16][256] xin f16 A-tile
  _Float16* zt = xa + 16 * 256;                 // [4][16][256] per-head nag
  float*    sc = (float*)(zt + 4 * 16 * 256);   // [16][4][8]
  float*    ss = sc + 16 * 4 * 8;               // [16][4]

  const int tid = threadIdx.x, wave = tid >> 5, lane = tid & 31;
  const int n0 = blockIdx.x * 16;
  const int h = wave;

  {  // async-gather 128 edge rows (1 thread each, 64 b128)
    int eidx = n2e[(size_t)(n0 + (tid >> 3)) * 8 + (tid & 7)];
    const float* src = edges + (size_t)eidx * 256;
    float* dst = eg + (size_t)tid * 256;
#pragma unroll 16
    for (int i = 0; i < 64; ++i) async_b128(dst + i * 4, src + i * 4);
  }
  {  // xin tile: 8 threads/row, 32 f32 (8 b128) each
    int i = tid >> 3, part = tid & 7;
    const float* src = xin + (size_t)(n0 + i) * ldx + part * 32;
    float* dst = xf + i * 256 + part * 32;
#pragma unroll
    for (int t = 0; t < 8; ++t) async_b128(dst + t * 4, src + t * 4);
  }
  wait_async0();
  __syncthreads();

  {  // build f16 copy of xin (WMMA A operand)
    int i = tid >> 3, part = tid & 7;
    const float* sf = xf + i * 256 + part * 32;
    _Float16* dh = xa + i * 256 + part * 32;
#pragma unroll
    for (int t = 0; t < 32; ++t) dh[t] = (_Float16)sf[t];
  }
  {  // attention dots: 128 tasks/wave -> 4 per lane
    const float* ae = a_ed + h * 256;
    for (int t = lane; t < 128; t += 32) {
      const float* row = eg + (size_t)t * 256;
      float s = 0.f;
      for (int d = 0; d < 256; ++d) s += row[d] * ae[d];
      sc[((t >> 3) * 4 + h) * 8 + (t & 7)] = s;
    }
  }
  if (lane < 16) {  // self dots
    const float* as = a_s + h * 256;
    const float* row = xf + lane * 256;
    float s = 0.f;
    for (int d = 0; d < 256; ++d) s += row[d] * as[d];
    ss[lane * 4 + h] = s;
  }
  __syncthreads();

  // softmax over K=8 and nag accumulation (f16 into z tile)
  for (int i = 0; i < 16; ++i) {
    float sb = ss[i * 4 + h];
    float l[8], m = -1e30f;
#pragma unroll
    for (int k = 0; k < 8; ++k) { l[k] = lrelu(sb + sc[(i * 4 + h) * 8 + k]); m = fmaxf(m, l[k]); }
    float sum = 0.f;
#pragma unroll
    for (int k = 0; k < 8; ++k) { l[k] = __expf(l[k] - m); sum += l[k]; }
    float inv = 1.f / sum;
#pragma unroll
    for (int k = 0; k < 8; ++k) l[k] *= inv;
    _Float16* zd = zt + (h * 16 + i) * 256;
#pragma unroll
    for (int j = 0; j < 8; ++j) {
      int d = lane + 32 * j;
      float acc = 0.f;
#pragma unroll
      for (int k = 0; k < 8; ++k) acc += l[k] * eg[(size_t)(i * 8 + k) * 256 + d];
      zd[d] = (_Float16)acc;
    }
  }
  __syncthreads();

  // dual-A projection: xin @ W_self[h] + nag @ W_nb[h], shared accumulators
  const float* Ws = Wself + (size_t)h * 256 * 64;
  const float* Wn = Wnb   + (size_t)h * 256 * 64;
  const _Float16* zh = zt + h * 16 * 256;
  v8f a0 = zero8(), a1 = zero8(), a2 = zero8(), a3 = zero8();
  for (int k0 = 0; k0 < 256; k0 += 32) {
    v16h ax = load_a_lds_f16(xa, 256, k0, lane);
    v16h az = load_a_lds_f16(zh, 256, k0, lane);
    a0 = wmma16(ax, load_b_glb(Ws, 64, k0,  0, lane), a0);
    a0 = wmma16(az, load_b_glb(Wn, 64, k0,  0, lane), a0);
    a1 = wmma16(ax, load_b_glb(Ws, 64, k0, 16, lane), a1);
    a1 = wmma16(az, load_b_glb(Wn, 64, k0, 16, lane), a1);
    a2 = wmma16(ax, load_b_glb(Ws, 64, k0, 32, lane), a2);
    a2 = wmma16(az, load_b_glb(Wn, 64, k0, 32, lane), a2);
    a3 = wmma16(ax, load_b_glb(Ws, 64, k0, 48, lane), a3);
    a3 = wmma16(az, load_b_glb(Wn, 64, k0, 48, lane), a3);
  }
  const int n = lane & 15, mb = (lane >> 4) * 8;
#pragma unroll
  for (int r = 0; r < 8; ++r) {
    float* orow = out + (size_t)(n0 + mb + r) * ldo + h * 64 + n;
    orow[0]  = eluf(a0[r]);
    orow[16] = eluf(a1[r]);
    orow[32] = eluf(a2[r]);
    orow[48] = eluf(a3[r]);
  }
}

// ====================================================================
extern "C" void kernel_launch(void* const* d_in, const int* in_sizes, int n_in,
                              void* d_out, int out_size, void* d_ws, size_t ws_size,
                              hipStream_t stream) {
  (void)in_sizes; (void)n_in; (void)out_size; (void)ws_size;
  const float* feats    = (const float*)d_in[0];
  const float* edge_emb = (const float*)d_in[1];
  const int*   ena      = (const int*)  d_in[2];
  const int*   n2e      = (const int*)  d_in[3];
  const float* W_prep0  = (const float*)d_in[4];
  const float* W_prep1  = (const float*)d_in[5];
  const float* W_eprep  = (const float*)d_in[6];
  const float* a_e      = (const float*)d_in[7];
  const float* a_n      = (const float*)d_in[8];
  const float* W_e      = (const float*)d_in[9];
  const float* a_s      = (const float*)d_in[10];
  const float* a_edge   = (const float*)d_in[11];
  const float* W_self   = (const float*)d_in[12];
  const float* W_nb     = (const float*)d_in[13];
  float* out = (float*)d_out;

  // workspace layout (fp32): dummy[N,D] | base[N,D] | edges0[E,D] | eout0[E,D]
  float* dummy  = (float*)d_ws;
  float* base   = dummy  + (size_t)N_ * D_;
  float* edges0 = base   + (size_t)N_ * D_;
  float* eout0  = edges0 + (size_t)E_ * D_;

  const dim3 blk(128);
  const size_t edge_smem =
      (32 * 256 + 16 * 256) * sizeof(float) + 4 * 16 * 256 * sizeof(_Float16) +
      16 * 4 * 3 * sizeof(float);                                    // ~81 KB
  const size_t node_smem =
      (16 * 8 * 256 + 16 * 256) * sizeof(float) +
      (16 * 256 + 4 * 16 * 256) * sizeof(_Float16) +
      (16 * 4 * 8 + 16 * 4) * sizeof(float);                         // ~186 KB

  // prep GEMMs: dummy = feats @ W_prep0 ; base = feats @ W_prep1
  gemm_f16_wmma<<<dim3(D_ / 64, N_ / 16), blk, 0, stream>>>(feats, D_, W_prep0, D_, dummy, D_, D_);
  gemm_f16_wmma<<<dim3(D_ / 64, N_ / 16), blk, 0, stream>>>(feats, D_, W_prep1, D_, base,  D_, D_);

  for (int mp = 0; mp < NMP_; ++mp) {
    // edges0 = edge_emb[mp] @ W_eprep[mp]
    gemm_f16_wmma<<<dim3(D_ / 64, E_ / 16), blk, 0, stream>>>(
        edge_emb + (size_t)mp * E_ * ED_, ED_,
        W_eprep  + (size_t)mp * ED_ * D_, D_, edges0, D_, ED_);

    const size_t off0  = (size_t)(mp * DEPTH_ + 0) * H_ * D_;
    const size_t offW0 = (size_t)(mp * DEPTH_ + 0) * H_ * D_ * PH_;
    const size_t off1  = (size_t)(mp * DEPTH_ + 1) * H_ * D_;
    const size_t offW1 = (size_t)(mp * DEPTH_ + 1) * H_ * D_ * PH_;
    float* out_mp = out + (size_t)mp * N_ * (DEPTH_ * D_);

    // L=0 edge aggregator (x = base) -> eout0
    edge_attn_proj<<<E_ / 16, blk, edge_smem, stream>>>(
        base, edges0, ena + (size_t)mp * E_ * 2,
        a_n + off0, a_e + off0, W_e + offW0, eout0);

    // L=0 node aggregator (xin = dummy, edges = edges0) -> out[:, 0:256]
    node_attn_proj<<<N_ / 16, blk, node_smem, stream>>>(
        dummy, D_, edges0, n2e + (size_t)mp * N_ * K_,
        a_s + off0, a_edge + off0, W_self + offW0, W_nb + offW0,
        out_mp, DEPTH_ * D_);

    // L=1 edge aggregator output is dead (never consumed) -> skipped.
    // L=1 node aggregator (xin = n_out0 from d_out, edges = eout0) -> out[:, 256:512]
    node_attn_proj<<<N_ / 16, blk, node_smem, stream>>>(
        out_mp, DEPTH_ * D_, eout0, n2e + (size_t)mp * N_ * K_,
        a_s + off1, a_edge + off1, W_self + offW1, W_nb + offW1,
        out_mp + D_, DEPTH_ * D_);
  }
}
